// GCNLayer_v1_11184094839116
// MI455X (gfx1250) — matrix operand order repaired and checked by reading, not compiled, not run on hardware
//
#include <hip/hip_runtime.h>
#include <hip/hip_bf16.h>
#include <math.h>

typedef __attribute__((ext_vector_type(16))) _Float16 v16h;
typedef __attribute__((ext_vector_type(8)))  float    v8f;

#define NN   100000
#define EE   1200000
#define INF  256
#define OUTF 64

// Load one 16x32 f16 A-fragment (ISA 16-bit A layout) from a row-major f32 row.
//   lane<16 : row M=lane,    K = {0..7, 16..23}
//   lane>=16: row M=lane-16, K = {8..15, 24..31}
// p = &x[row_of_this_lane][kstep*32 + (lane>=16 ? 8 : 0)]
__device__ __forceinline__ v16h load_a_frag(const float* __restrict__ p) {
  float4 f0 = *(const float4*)(p + 0);
  float4 f1 = *(const float4*)(p + 4);
  float4 f2 = *(const float4*)(p + 16);
  float4 f3 = *(const float4*)(p + 20);
  v16h a;
  a[0]  = (_Float16)f0.x; a[1]  = (_Float16)f0.y;
  a[2]  = (_Float16)f0.z; a[3]  = (_Float16)f0.w;
  a[4]  = (_Float16)f1.x; a[5]  = (_Float16)f1.y;
  a[6]  = (_Float16)f1.z; a[7]  = (_Float16)f1.w;
  a[8]  = (_Float16)f2.x; a[9]  = (_Float16)f2.y;
  a[10] = (_Float16)f2.z; a[11] = (_Float16)f2.w;
  a[12] = (_Float16)f3.x; a[13] = (_Float16)f3.y;
  a[14] = (_Float16)f3.z; a[15] = (_Float16)f3.w;
  return a;
}

// ---------------------------------------------------------------------------
// Kernel 1: support = X @ W via v_wmma_f32_16x16x32_f16
//   block = 256 threads = 8 waves; each wave: 32 rows x 64 cols (2 M-tiles x
//   4 N-tiles) -> every LDS B-fragment feeds 2 WMMAs.
//   W (256x64 f32) packed once per block into LDS as f16 B-fragments.
// LDS B layout: frag(ks in 0..7, nt in 0..3) : [lane 0..31][16 halves]
//   lane = (n%16) + 16*(((k%32)>>3)&1), half h = (k%8) + 8*((k%32)>=16)  (ISA 16-bit B layout)
// ---------------------------------------------------------------------------
__global__ __launch_bounds__(256) void gcn_gemm_wmma(
    const float* __restrict__ x, const float* __restrict__ w,
    float* __restrict__ support) {
  __shared__ __align__(32) _Float16 Bp[8 * 4 * 32 * 16];  // 32 KB

  // cooperative pack: W[k][n] f32 -> f16 fragment layout
  for (int i = threadIdx.x; i < INF * OUTF; i += 256) {
    int k = i >> 6, n = i & 63;
    int ks = k >> 5, kr = k & 31;
    int nt = n >> 4, nl = n & 15;
    int lane = nl + ((kr & 8) << 1);    // +16 if (kr & 8)
    int h    = (kr & 7) | ((kr >> 1) & 8);
    Bp[((((ks << 2) + nt) << 5) + lane) * 16 + h] = (_Float16)w[i];
  }
  __syncthreads();

  const int wave = threadIdx.x >> 5;
  const int lane = threadIdx.x & 31;
  const long baseRow = ((long)blockIdx.x * 8 + wave) * 32;   // 100000 = 3125*32
  if (baseRow >= NN) return;            // wave-uniform: EXEC stays all-ones

  const int m  = lane & 15;
  const int c0 = (lane >= 16) ? 8 : 0;
  const float* xrow0 = x + (baseRow + m) * INF + c0;         // M-tile 0
  const float* xrow1 = x + (baseRow + 16 + m) * INF + c0;    // M-tile 1

  v8f acc[8];                            // acc[mt*4 + nt]
#pragma unroll
  for (int i = 0; i < 8; ++i) acc[i] = (v8f){};

#pragma unroll
  for (int ks = 0; ks < 8; ++ks) {
    const float* p0 = xrow0 + ks * 32;
    const float* p1 = xrow1 + ks * 32;
    if (ks < 7) {                        // WGP-scope streaming prefetch
      __builtin_prefetch(p0 + 32, 0, 3);
      __builtin_prefetch(p1 + 32, 0, 3);
    }
    v16h a0 = load_a_frag(p0);
    v16h a1 = load_a_frag(p1);

    const v16h* bbase = ((const v16h*)Bp) + ((ks << 2) << 5) + lane;
#pragma unroll
    for (int nt = 0; nt < 4; ++nt) {
      v16h b = bbase[nt * 32];
      acc[nt]     = __builtin_amdgcn_wmma_f32_16x16x32_f16(
          false, a0, false, b, (short)0, acc[nt],     false, false);
      acc[4 + nt] = __builtin_amdgcn_wmma_f32_16x16x32_f16(
          false, a1, false, b, (short)0, acc[4 + nt], false, false);
    }
  }

  // C/D layout: VGPR r, lane l -> (M = r + 8*(l>>4), N = l&15)
  const int colBase = lane & 15;
  const int rowOff  = (lane >> 4) * 8;
#pragma unroll
  for (int mt = 0; mt < 2; ++mt) {
#pragma unroll
    for (int r = 0; r < 8; ++r) {
      float* q = support + (baseRow + mt * 16 + rowOff + r) * OUTF + colBase;
      q[0]  = acc[mt * 4 + 0][r];
      q[16] = acc[mt * 4 + 1][r];
      q[32] = acc[mt * 4 + 2][r];
      q[48] = acc[mt * 4 + 3][r];
    }
  }
}

// ---------------------------------------------------------------------------
// Kernel 2: zero the accumulator (d_out) -- graph-replay-safe re-init
// ---------------------------------------------------------------------------
__global__ __launch_bounds__(256) void gcn_zero(float4* __restrict__ out) {
  long i = (long)blockIdx.x * 256 + threadIdx.x;   // NN*OUTF/4 float4s
  if (i < (long)NN * OUTF / 4) out[i] = make_float4(0.f, 0.f, 0.f, 0.f);
}

// ---------------------------------------------------------------------------
// Kernel 3: edge scatter  out[dst] += ew * support[src]
//   16 lanes per edge, 4 cols (float4) per lane -> coalesced 256B gather/edge
// ---------------------------------------------------------------------------
__global__ __launch_bounds__(256) void gcn_scatter(
    const int* __restrict__ src, const int* __restrict__ dst,
    const float* __restrict__ ew, const float* __restrict__ support,
    float* __restrict__ out) {
  long gid = (long)blockIdx.x * 256 + threadIdx.x;
  long e = gid >> 4;
  if (e >= EE) return;
  int c4 = (int)(gid & 15);

  int   s  = src[e];
  int   d  = dst[e];
  float wv = ew[e];

  float4 v = *((const float4*)(support + (long)s * OUTF) + c4);
  float* op = out + (long)d * OUTF + c4 * 4;
  atomicAdd(op + 0, wv * v.x);
  atomicAdd(op + 1, wv * v.y);
  atomicAdd(op + 2, wv * v.z);
  atomicAdd(op + 3, wv * v.w);
}

// ---------------------------------------------------------------------------
// Kernel 4: finalize  out = sigmoid(out / norm + b)   (in-place, elementwise)
// ---------------------------------------------------------------------------
__global__ __launch_bounds__(256) void gcn_finalize(
    float* __restrict__ out, const float* __restrict__ norm,
    const float* __restrict__ b) {
  long i = (long)blockIdx.x * 256 + threadIdx.x;
  if (i >= (long)NN * OUTF) return;
  long row = i >> 6;
  int  col = (int)(i & 63);
  float v = out[i] / norm[row] + b[col];
  out[i] = 1.0f / (1.0f + __expf(-v));
}

extern "C" void kernel_launch(void* const* d_in, const int* in_sizes, int n_in,
                              void* d_out, int out_size, void* d_ws, size_t ws_size,
                              hipStream_t stream) {
  const float* x    = (const float*)d_in[0];
  const int*   esrc = (const int*)d_in[1];
  const int*   edst = (const int*)d_in[2];
  const float* ew   = (const float*)d_in[3];
  const float* norm = (const float*)d_in[4];
  const float* w    = (const float*)d_in[5];
  const float* b    = (const float*)d_in[6];
  float* out     = (float*)d_out;
  float* support = (float*)d_ws;       // NN*OUTF floats = 25.6 MB scratch

  // 1) zero accumulator
  {
    long n4 = (long)NN * OUTF / 4;
    int blocks = (int)((n4 + 255) / 256);
    gcn_zero<<<blocks, 256, 0, stream>>>((float4*)out);
  }
  // 2) dense GEMM via WMMA: 256 rows / block (8 waves x 32 rows)
  {
    int blocks = (NN + 255) / 256;
    gcn_gemm_wmma<<<blocks, 256, 0, stream>>>(x, w, support);
  }
  // 3) edge scatter (atomic segment-sum)
  {
    long threads = (long)EE * 16;
    int blocks = (int)((threads + 255) / 256);
    gcn_scatter<<<blocks, 256, 0, stream>>>(esrc, edst, ew, support, out);
  }
  // 4) finalize: /norm + bias, sigmoid
  {
    long threads = (long)NN * OUTF;
    int blocks = (int)((threads + 255) / 256);
    gcn_finalize<<<blocks, 256, 0, stream>>>(out, norm, b);
  }
}
